// GGNNSum_26886495273788
// MI455X (gfx1250) — matrix operand order repaired and checked by reading, not compiled, not run on hardware
//
#include <hip/hip_runtime.h>
#include <hip/hip_bf16.h>
#include <stdint.h>

#define DD      256
#define IN_DIMM 128
#define TET     4
#define NSTEP   8

typedef __attribute__((ext_vector_type(16))) __bf16 v16bf;
typedef __attribute__((ext_vector_type(8)))  float  v8f;

union FragBF { v16bf v; uint4 q[2]; };

__device__ __forceinline__ unsigned short f2bf(float f) {
    unsigned int u = __float_as_uint(f);
    if ((u & 0x7fffffffu) > 0x7f800000u) return (unsigned short)((u >> 16) | 0x40); // NaN
    u += 0x7fffu + ((u >> 16) & 1u);   // round-to-nearest-even
    return (unsigned short)(u >> 16);
}
__device__ __forceinline__ float bf2f(unsigned int h) { return __uint_as_float(h << 16); }

// ---------------- utility kernels ----------------

__global__ void fzero(float* p, size_t n) {
    size_t i = (size_t)blockIdx.x * blockDim.x + threadIdx.x;
    size_t st = (size_t)gridDim.x * blockDim.x;
    for (; i < n; i += st) p[i] = 0.f;
}

__global__ void cvt_f2bf(const float* __restrict__ s, unsigned short* __restrict__ d, size_t n) {
    size_t i = (size_t)blockIdx.x * blockDim.x + threadIdx.x;
    size_t st = (size_t)gridDim.x * blockDim.x;
    for (; i < n; i += st) d[i] = f2bf(s[i]);
}

// Wt[t][e][d] = bf16(W[t][d][e])  (pre-transposed so B-columns are contiguous along K)
__global__ void prep_wt(const float* __restrict__ W, unsigned short* __restrict__ Wt) {
    int idx = blockIdx.x * 256 + threadIdx.x;      // t*65536 + e*256 + d
    int t = idx >> 16, e = (idx >> 8) & 255, d = idx & 255;
    Wt[idx] = f2bf(W[(t << 16) | (d << 8) | e]);
}

// h = [features | zeros], plus bf16 mirror for WMMA A operand
__global__ void init_h(const float* __restrict__ feat, float* __restrict__ h,
                       unsigned short* __restrict__ hb, int Nn) {
    size_t idx = (size_t)blockIdx.x * blockDim.x + threadIdx.x;
    if (idx >= (size_t)Nn * DD) return;
    int d = (int)(idx & (DD - 1));
    size_t n = idx >> 8;
    float v = (d < IN_DIMM) ? feat[n * IN_DIMM + d] : 0.f;
    h[idx] = v;
    hb[idx] = f2bf(v);
}

// ---------------- WMMA GEMM:  C[m,n] = sum_k A[m,k]*B[n,k] + bias[n] ----------------
// A: [M,K] bf16 row-major, B: [Ncols,K] bf16 ("B^T" rows = contiguous K slices)
// Each wave computes a 16x32 tile (two 16x16 WMMA accumulators sharing one A fragment).
template <bool OUT_BF16>
__global__ __launch_bounds__(256)
void wmma_gemm(const unsigned short* __restrict__ A,
               const unsigned short* __restrict__ B,
               void* __restrict__ Cv,
               const float* __restrict__ bias,
               int M, int K, int ldc)
{
    const int lane = threadIdx.x & 31;
    const int wave = threadIdx.x >> 5;                 // 0..7
    const int m0 = blockIdx.x * 128 + wave * 16;
    const int n0 = blockIdx.y * 32;
    if (m0 >= M) return;                               // uniform per-wave exit (EXEC stays all-1 for WMMA)

    const int half = lane >> 4;                        // 0: low lanes, 1: high lanes
    const int l16  = lane & 15;

    int arow = m0 + l16; if (arow >= M) arow = M - 1;  // clamp loads; stores guarded below
    const unsigned short* Ap  = A + (size_t)arow * K;
    const unsigned short* Bp0 = B + (size_t)(n0 + l16) * K;
    const unsigned short* Bp1 = B + (size_t)(n0 + 16 + l16) * K;

    v8f acc0 = {}, acc1 = {};
    #pragma unroll
    for (int k0 = 0; k0 < 256; k0 += 32) {
        FragBF fa, fb0, fb1;
        // A 16x32 bf16 lane layout: lanes 0-15 K {k0..k0+15}, lanes 16-31 K {k0+16..k0+31}
        fa.q[0]  = *(const uint4*)(Ap  + k0 + half * 16);
        fa.q[1]  = *(const uint4*)(Ap  + k0 + half * 16 + 8);
        // B 32x16 bf16 lane layout: lanes 0-15 K {k0..k0+15}, lanes 16-31 K {k0+16..k0+31}
        fb0.q[0] = *(const uint4*)(Bp0 + k0 + half * 16);
        fb0.q[1] = *(const uint4*)(Bp0 + k0 + half * 16 + 8);
        fb1.q[0] = *(const uint4*)(Bp1 + k0 + half * 16);
        fb1.q[1] = *(const uint4*)(Bp1 + k0 + half * 16 + 8);
        acc0 = __builtin_amdgcn_wmma_f32_16x16x32_bf16(false, fa.v, false, fb0.v,
                                                       (short)0, acc0, false, false);
        acc1 = __builtin_amdgcn_wmma_f32_16x16x32_bf16(false, fa.v, false, fb1.v,
                                                       (short)0, acc1, false, false);
    }

    // C/D layout: VGPR i -> row m0 + half*8 + i, column n0 + (lane%16)
    const int col0 = n0 + l16;
    const int col1 = col0 + 16;
    const float b0 = bias[col0];
    const float b1 = bias[col1];
    #pragma unroll
    for (int i = 0; i < 8; ++i) {
        int r = m0 + half * 8 + i;
        if (r < M) {
            float v0 = acc0[i] + b0;
            float v1 = acc1[i] + b1;
            if constexpr (OUT_BF16) {
                unsigned short* C = (unsigned short*)Cv;
                C[(size_t)r * ldc + col0] = f2bf(v0);
                C[(size_t)r * ldc + col1] = f2bf(v1);
            } else {
                float* C = (float*)Cv;
                C[(size_t)r * ldc + col0] = v0;
                C[(size_t)r * ldc + col1] = v1;
            }
        }
    }
}

// ---------------- edge message gather + segment-sum (fp32 atomics) ----------------
// 64 lanes per edge, 4 dims per lane (bf16 Wh keeps gather mostly L2-resident)
__global__ void edge_scatter(const unsigned short* __restrict__ Wh,
                             const int* __restrict__ esrc, const int* __restrict__ edst,
                             const int* __restrict__ etyp, float* __restrict__ agg,
                             int E, int Nn)
{
    int e = blockIdx.x * 4 + (threadIdx.x >> 6);
    if (e >= E) return;
    int d = (threadIdx.x & 63) << 2;
    int s = esrc[e], t = etyp[e], dd = edst[e];
    const unsigned short* mp = Wh + (((size_t)t * Nn + s) << 8) + d;
    uint2 m4 = *(const uint2*)mp;
    float* out = agg + ((size_t)dd << 8) + d;
    atomicAdd(out + 0, bf2f(m4.x & 0xffffu));
    atomicAdd(out + 1, bf2f(m4.x >> 16));
    atomicAdd(out + 2, bf2f(m4.y & 0xffffu));
    atomicAdd(out + 3, bf2f(m4.y >> 16));
}

// ---------------- fused GRU gates; writes h (fp32) and bf16 mirror ----------------
__global__ void gru_gates(const float* __restrict__ gi, const float* __restrict__ gh,
                          float* __restrict__ h, unsigned short* __restrict__ hb, int Nn)
{
    size_t idx = (size_t)blockIdx.x * blockDim.x + threadIdx.x;
    if (idx >= (size_t)Nn * DD) return;
    size_t n = idx >> 8;
    int d = (int)(idx & (DD - 1));
    const float* gip = gi + n * (3 * DD);
    const float* ghp = gh + n * (3 * DD);
    float r  = 1.f / (1.f + __expf(-(gip[d]          + ghp[d])));
    float z  = 1.f / (1.f + __expf(-(gip[DD + d]     + ghp[DD + d])));
    float nn = tanhf(gip[2 * DD + d] + r * ghp[2 * DD + d]);
    float hv = h[idx];
    float o  = (1.f - z) * nn + z * hv;
    h[idx]  = o;
    hb[idx] = f2bf(o);
}

// ---------------- sum-pool + classifier ----------------
__global__ void pool_sum(const float* __restrict__ h, float* __restrict__ pooled, int Nn) {
    int d = threadIdx.x;                         // 256 dims
    float s = 0.f;
    for (int n = blockIdx.x; n < Nn; n += gridDim.x) s += h[(size_t)n * DD + d];
    atomicAdd(&pooled[d], s);
}

__global__ void classify(const float* __restrict__ pooled, const float* __restrict__ Wc,
                         const float* __restrict__ bc, float* __restrict__ out) {
    __shared__ float red[DD];
    int d = threadIdx.x;
    red[d] = pooled[d] * Wc[d];
    __syncthreads();
    for (int s = 128; s > 0; s >>= 1) { if (d < s) red[d] += red[d + s]; __syncthreads(); }
    if (d == 0) out[0] = 1.f / (1.f + __expf(-(red[0] + bc[0])));
}

// ---------------- host launcher ----------------
extern "C" void kernel_launch(void* const* d_in, const int* in_sizes, int n_in,
                              void* d_out, int out_size, void* d_ws, size_t ws_size,
                              hipStream_t stream)
{
    const float* feat = (const float*)d_in[0];
    const int*   esrc = (const int*)  d_in[1];
    const int*   edst = (const int*)  d_in[2];
    const int*   etyp = (const int*)  d_in[3];
    const float* W    = (const float*)d_in[4];
    const float* b    = (const float*)d_in[5];
    const float* Wih  = (const float*)d_in[6];
    const float* Whh  = (const float*)d_in[7];
    const float* bih  = (const float*)d_in[8];
    const float* bhh  = (const float*)d_in[9];
    const float* Wc   = (const float*)d_in[10];
    const float* bc   = (const float*)d_in[11];
    float* out = (float*)d_out;

    const int Nn = in_sizes[0] / IN_DIMM;   // 50000
    const int E  = in_sizes[1];             // 800000
    const size_t ND = (size_t)Nn * DD;

    char* ws = (char*)d_ws;
    size_t off = 0;
    auto alloc = [&](size_t bytes) -> void* {
        off = (off + 255) & ~(size_t)255;
        void* p = ws + off;
        off += bytes;
        return p;
    };
    float*          h      = (float*)         alloc(ND * 4);
    unsigned short* hb     = (unsigned short*)alloc(ND * 2);
    float*          agg    = (float*)         alloc(ND * 4);
    unsigned short* aggb   = (unsigned short*)alloc(ND * 2);
    unsigned short* Wt     = (unsigned short*)alloc((size_t)TET * DD * DD * 2);
    unsigned short* Wihb   = (unsigned short*)alloc((size_t)3 * DD * DD * 2);
    unsigned short* Whhb   = (unsigned short*)alloc((size_t)3 * DD * DD * 2);
    float*          pooled = (float*)         alloc(DD * 4);
    // scratch union: Wh (bf16, T*N*D) is dead before gi/gh are written
    off = (off + 255) & ~(size_t)255;
    unsigned short* Wh = (unsigned short*)(ws + off);          // TET*ND*2 bytes
    float*          gi = (float*)(ws + off);                    // ND*3*4 bytes (overlaps Wh)
    float*          gh = (float*)(ws + off + ND * 3 * 4);       // ND*3*4 bytes

    // --- one-time prep (re-run each call: deterministic, inputs never mutated) ---
    prep_wt<<<(TET * DD * DD) / 256, 256, 0, stream>>>(W, Wt);
    cvt_f2bf<<<768, 256, 0, stream>>>(Wih, Wihb, (size_t)3 * DD * DD);
    cvt_f2bf<<<768, 256, 0, stream>>>(Whh, Whhb, (size_t)3 * DD * DD);
    init_h<<<(int)((ND + 255) / 256), 256, 0, stream>>>(feat, h, hb, Nn);

    const dim3 gD((Nn + 127) / 128, DD / 32);        // Ncols = 256
    const dim3 g3D((Nn + 127) / 128, (3 * DD) / 32); // Ncols = 768

    for (int step = 0; step < NSTEP; ++step) {
        // Wh[t] = h @ W[t] + b[t]  -> bf16 (kept hot in L2 for the gather)
        for (int t = 0; t < TET; ++t)
            wmma_gemm<true><<<gD, 256, 0, stream>>>(
                hb, Wt + (size_t)t * DD * DD, (void*)(Wh + (size_t)t * ND),
                b + t * DD, Nn, DD, DD);

        fzero<<<2048, 256, 0, stream>>>(agg, ND);
        edge_scatter<<<(E + 3) / 4, 256, 0, stream>>>(Wh, esrc, edst, etyp, agg, E, Nn);
        cvt_f2bf<<<2048, 256, 0, stream>>>(agg, aggb, ND);

        // gi = agg @ W_ih^T + b_ih ; gh = h @ W_hh^T + b_hh   (fp32 outputs)
        wmma_gemm<false><<<g3D, 256, 0, stream>>>(aggb, Wihb, (void*)gi, bih, Nn, DD, 3 * DD);
        wmma_gemm<false><<<g3D, 256, 0, stream>>>(hb,   Whhb, (void*)gh, bhh, Nn, DD, 3 * DD);

        gru_gates<<<(int)((ND + 255) / 256), 256, 0, stream>>>(gi, gh, h, hb, Nn);
    }

    fzero<<<1, 256, 0, stream>>>(pooled, DD);
    pool_sum<<<256, 256, 0, stream>>>(h, pooled, Nn);
    classify<<<1, 256, 0, stream>>>(pooled, Wc, bc, out);
}
